// QWen3Attention_6682969113250
// MI455X (gfx1250) — compile-verified, matrix operand order checked by reading, not run on hardware
//
#include <hip/hip_runtime.h>
#include <hip/hip_bf16.h>

// ---------------- types ----------------
typedef __bf16 bf16;
typedef __attribute__((ext_vector_type(16))) __bf16 v16bf;
typedef __attribute__((ext_vector_type(8)))  __bf16 v8bf;
typedef __attribute__((ext_vector_type(8)))  float  v8f;

union V16 { v16bf v; v8bf h[2]; };

// ---------------- problem constants ----------------
#define B_    2
#define S_    2048
#define HID_  4096
#define H_    32
#define KVH_  8
#define D_    128
#define BS_   (B_*S_)
#define EPS_  1e-6f
#define LOG_BASE 13.815510557964274f     // ln(1e6)
#define SCALE_   0.08838834764831845f    // 1/sqrt(128)

__device__ __forceinline__ bf16 f2bf(float f) {
  unsigned u = __builtin_bit_cast(unsigned, f);
  u += 0x7FFFu + ((u >> 16) & 1u);                 // round-to-nearest-even
  unsigned short h = (unsigned short)(u >> 16);
  return __builtin_bit_cast(bf16, h);
}

// LDS byte address (what DS / async-to-LDS instructions consume)
__device__ __forceinline__ unsigned lds_off(void* p) {
  return (unsigned)(unsigned long long)(__attribute__((address_space(3))) char*)p;
}

// CDNA5 async global->LDS DMA (ASYNCcnt-tracked), 16B per lane
__device__ __forceinline__ void async_copy_b128(unsigned lds_dst, const void* gsrc) {
  asm volatile("global_load_async_to_lds_b128 %0, %1, off"
               :: "v"(lds_dst), "v"((unsigned long long)(size_t)gsrc)
               : "memory");
}
__device__ __forceinline__ void wait_async0() {
  asm volatile("s_wait_asynccnt 0" ::: "memory");
}

// ---------------- kernel 0a: f32 -> bf16 ----------------
__global__ __launch_bounds__(256) void f32_to_bf16_kernel(
    const float* __restrict__ src, bf16* __restrict__ dst, int n) {
  int i = (blockIdx.x * 256 + threadIdx.x) * 4;
  if (i < n) {
    float4 x = *(const float4*)(src + i);
    dst[i + 0] = f2bf(x.x); dst[i + 1] = f2bf(x.y);
    dst[i + 2] = f2bf(x.z); dst[i + 3] = f2bf(x.w);
  }
}

// ---------------- kernel 0b: LDS-tiled transpose f32[R,C] -> bf16[C,R] ----------
// 32x32 tiles: coalesced reads along C, coalesced writes along R.
__global__ __launch_bounds__(256) void transpose_to_bf16_kernel(
    const float* __restrict__ src, bf16* __restrict__ dst, int R, int Cn) {
  __shared__ float ls[32][33];
  const int tiles_c = Cn >> 5;
  const int tr = blockIdx.x / tiles_c;      // tile along R (k)
  const int tc = blockIdx.x % tiles_c;      // tile along C (n)
  const int k0 = tr * 32, n0 = tc * 32;
  const int tid = threadIdx.x;
#pragma unroll
  for (int i = 0; i < 4; i++) {
    int idx = tid + i * 256;
    int r = idx >> 5, c = idx & 31;
    ls[r][c] = src[(size_t)(k0 + r) * Cn + n0 + c];
  }
  __syncthreads();
#pragma unroll
  for (int i = 0; i < 4; i++) {
    int idx = tid + i * 256;
    int r = idx >> 5, c = idx & 31;
    dst[(size_t)(n0 + r) * R + k0 + c] = f2bf(ls[c][r]);
  }
}

// ---------------- kernel 0c: V pack: copy f32 out + LDS-tiled transpose -------
// Vf [BS, KVH*D] -> v_out (f32, same layout, coalesced) and Vt [B,KVH,D,S] bf16.
__global__ __launch_bounds__(256) void v_pack_transpose_kernel(
    const float* __restrict__ Vf, float* __restrict__ v_out,
    bf16* __restrict__ Vt) {
  __shared__ float ls[32][33];
  const int bg = blockIdx.y;               // b*KVH + kvh
  const int b = bg / KVH_, kvh = bg % KVH_;
  const int tiles_d = D_ >> 5;             // 4
  const int ts = blockIdx.x / tiles_d;     // tile along S
  const int td = blockIdx.x % tiles_d;     // tile along D
  const int s0 = ts * 32, d0 = td * 32;
  const int tid = threadIdx.x;
#pragma unroll
  for (int i = 0; i < 4; i++) {
    int idx = tid + i * 256;
    int r = idx >> 5, c = idx & 31;        // r: s, c: d
    size_t off = (size_t)(b * S_ + s0 + r) * (KVH_ * D_) + kvh * D_ + d0 + c;
    float x = Vf[off];
    ls[r][c] = x;
    v_out[off] = x;
  }
  __syncthreads();
#pragma unroll
  for (int i = 0; i < 4; i++) {
    int idx = tid + i * 256;
    int r = idx >> 5, c = idx & 31;        // r: d, c: s
    Vt[((size_t)bg * D_ + d0 + r) * S_ + s0 + c] = f2bf(ls[c][r]);
  }
}

// ---------------- kernel 1: NT bf16 GEMM, f32 out ----------------
// C[M,N] = A[M,K] * BT[N,K]^T.  256x128 block tile, TK=32, 8 waves in a
// 4x2 grid; each wave owns a 64x64 region = 4x4 wmma tiles.  Tiles staged
// with async global->LDS DMA, double buffered, one barrier per k-step.
#define TM_ 256
#define TN_ 128
#define TK_ 32
#define LSTR 40   // LDS row stride in halfs (32 + 8 pad, keeps 16B alignment)

__global__ __launch_bounds__(256) void gemm_nt_bf16(
    const bf16* __restrict__ A, const bf16* __restrict__ BT,
    float* __restrict__ C, int M, int N, int K) {
  __shared__ bf16 lA[2][TM_ * LSTR];
  __shared__ bf16 lB[2][TN_ * LSTR];
  const int tid  = threadIdx.x;
  const int lane = tid & 31;
  const int wave = tid >> 5;
  const int wm   = wave & 3;      // 4 waves along M (64 rows each)
  const int wn   = wave >> 2;     // 2 waves along N (64 cols each)
  const int hi   = lane >> 4;
  const int l16  = lane & 15;
  const int bm   = blockIdx.y * TM_;
  const int bn   = blockIdx.x * TN_;

  const v8f vz = {0.f,0.f,0.f,0.f,0.f,0.f,0.f,0.f};
  v8f acc[4][4];
#pragma unroll
  for (int i = 0; i < 4; i++)
#pragma unroll
    for (int j = 0; j < 4; j++) acc[i][j] = vz;

  auto stage = [&](int k0, int buf) {
#pragma unroll
    for (int i = 0; i < 4; i++) {            // A: 256 rows x 4 x 16B
      int idx = tid + i * 256;
      int row = idx >> 2, seg = (idx & 3) * 8;
      async_copy_b128(lds_off(&lA[buf][row * LSTR + seg]),
                      &A[(size_t)(bm + row) * K + k0 + seg]);
    }
#pragma unroll
    for (int i = 0; i < 2; i++) {            // B: 128 rows x 4 x 16B
      int idx = tid + i * 256;
      int row = idx >> 2, seg = (idx & 3) * 8;
      async_copy_b128(lds_off(&lB[buf][row * LSTR + seg]),
                      &BT[(size_t)(bn + row) * K + k0 + seg]);
    }
  };

  stage(0, 0);
  const int nsteps = K / TK_;
  for (int t = 0; t < nsteps; t++) {
    const int cur = t & 1;
    wait_async0();            // my async DMAs (tile t) have landed in LDS
    __syncthreads();          // tile t visible to all; buffer !cur reusable

    if (t + 1 < nsteps) stage((t + 1) * TK_, cur ^ 1);
    if ((t + 2) * TK_ < K) {  // speculative prefetch of tile t+2 into L2
      __builtin_prefetch(&A[(size_t)(bm + tid) * K + (t + 2) * TK_], 0, 0);
      if (tid < TN_)
        __builtin_prefetch(&BT[(size_t)(bn + tid) * K + (t + 2) * TK_], 0, 0);
    }

    V16 af[4], bfr[4];
#pragma unroll
    for (int tm = 0; tm < 4; tm++) {
      const bf16* p = &lA[cur][(wm * 64 + tm * 16 + l16) * LSTR];
      af[tm].h[0] = *(const v8bf*)(p + hi * 8);
      af[tm].h[1] = *(const v8bf*)(p + 16 + hi * 8);
    }
#pragma unroll
    for (int tn = 0; tn < 4; tn++) {
      const bf16* p = &lB[cur][(wn * 64 + tn * 16 + l16) * LSTR];
      bfr[tn].h[0] = *(const v8bf*)(p + hi * 8);
      bfr[tn].h[1] = *(const v8bf*)(p + 16 + hi * 8);
    }
#pragma unroll
    for (int tm = 0; tm < 4; tm++)
#pragma unroll
      for (int tn = 0; tn < 4; tn++)
        acc[tm][tn] = __builtin_amdgcn_wmma_f32_16x16x32_bf16(
            false, af[tm].v, false, bfr[tn].v, (short)0, acc[tm][tn], false, false);
  }

#pragma unroll
  for (int tm = 0; tm < 4; tm++)
#pragma unroll
    for (int tn = 0; tn < 4; tn++) {
      int r0 = bm + wm * 64 + tm * 16 + hi * 8;
      int c0 = bn + wn * 64 + tn * 16 + l16;
#pragma unroll
      for (int r = 0; r < 8; r++)
        C[(size_t)(r0 + r) * N + c0] = acc[tm][tn][r];
    }
}

// ---------------- kernel 2: RMSNorm + interleaved RoPE (q,k heads only) -------
__global__ __launch_bounds__(256) void rmsnorm_rope_pack(
    const int*   __restrict__ positions,
    const float* __restrict__ Qf,   // [BS, H*D]
    const float* __restrict__ Kf,   // [BS, KVH*D]
    const float* __restrict__ qw, const float* __restrict__ kw,
    bf16* __restrict__ Qb,          // [B,H,S,D]
    bf16* __restrict__ Kb,          // [B,KVH,S,D]
    float* __restrict__ k_out) {    // [BS, KVH*D]
  const int bs = blockIdx.x;
  const int b = bs / S_, s = bs % S_;
  const int wave = threadIdx.x >> 5, lane = threadIdx.x & 31;
  const float pos = (float)positions[s];
  const int d0 = lane * 4;
  const float i0 = (float)(d0 >> 1), i1 = i0 + 1.0f;
  const float f0 = pos * __expf(-(2.0f * i0 / D_) * LOG_BASE);
  const float f1 = pos * __expf(-(2.0f * i1 / D_) * LOG_BASE);
  float sn0, cs0, sn1, cs1;
  __sincosf(f0, &sn0, &cs0);
  __sincosf(f1, &sn1, &cs1);

  for (int t = wave; t < H_ + KVH_; t += 8) {   // 40 tasks / 8 waves
    const bool isq = (t < H_);
    const int h = isq ? t : (t - H_);
    const float* src = isq ? &Qf[(size_t)bs * (H_ * D_) + h * D_ + d0]
                           : &Kf[(size_t)bs * (KVH_ * D_) + h * D_ + d0];
    const float4 x  = *(const float4*)src;
    const float4 w4 = *(const float4*)((isq ? qw : kw) + d0);
    float ss = x.x * x.x + x.y * x.y + x.z * x.z + x.w * x.w;
#pragma unroll
    for (int m = 16; m >= 1; m >>= 1) ss += __shfl_xor(ss, m, 32);
    const float r = rsqrtf(ss * (1.0f / D_) + EPS_);
    const float y0 = x.x * r * w4.x, y1 = x.y * r * w4.y;
    const float y2 = x.z * r * w4.z, y3 = x.w * r * w4.w;
    const float o0 = y0 * cs0 - y1 * sn0, o1 = y1 * cs0 + y0 * sn0;
    const float o2 = y2 * cs1 - y3 * sn1, o3 = y3 * cs1 + y2 * sn1;
    if (isq) {
      bf16* dst = &Qb[(((size_t)b * H_ + h) * S_ + s) * D_ + d0];
      dst[0] = f2bf(o0); dst[1] = f2bf(o1); dst[2] = f2bf(o2); dst[3] = f2bf(o3);
    } else {
      bf16* dst = &Kb[(((size_t)b * KVH_ + h) * S_ + s) * D_ + d0];
      dst[0] = f2bf(o0); dst[1] = f2bf(o1); dst[2] = f2bf(o2); dst[3] = f2bf(o3);
      float* ko = &k_out[(size_t)bs * (KVH_ * D_) + h * D_ + d0];
      ko[0] = o0; ko[1] = o1; ko[2] = o2; ko[3] = o3;
    }
  }
}

// ---------------- kernel 3: flash attention ----------------
// 128 q rows/block, 8 waves x 16 rows; K/V^T tiles (64 keys) staged by
// async global->LDS DMA, double buffered, one barrier per k-tile.
#define QT_ 128
#define KT_ 64
#define KSTR (D_ + 8)    // 136 halfs
#define VSTR (KT_ + 8)   // 72 halfs

__global__ __launch_bounds__(256) void flash_attn_kernel(
    const bf16* __restrict__ Qb,   // [B,H,S,D]
    const bf16* __restrict__ Kb,   // [B,KVH,S,D]
    const bf16* __restrict__ Vt,   // [B,KVH,D,S]
    bf16* __restrict__ Ab) {       // [BS, H*D]
  __shared__ bf16 lK[2][KT_ * KSTR];
  __shared__ bf16 lV[2][D_ * VSTR];
  __shared__ bf16 lP[8][16 * VSTR];

  const int tid = threadIdx.x, lane = tid & 31, wave = tid >> 5;
  const int hi = lane >> 4, l16 = lane & 15;
  const int bh = blockIdx.y;
  const int b = bh / H_, h = bh % H_;
  const int g = h / (H_ / KVH_);
  const int q0 = blockIdx.x * QT_;

  const bf16* Qbase = Qb + ((size_t)b * H_ + h) * S_ * D_;
  const bf16* Kbase = Kb + ((size_t)b * KVH_ + g) * S_ * D_;
  const bf16* Vbase = Vt + ((size_t)b * KVH_ + g) * D_ * S_;

  V16 qf[4];
  {
    const bf16* qp = Qbase + (size_t)(q0 + wave * 16 + l16) * D_;
#pragma unroll
    for (int dk = 0; dk < 4; dk++) {
      qf[dk].h[0] = *(const v8bf*)(qp + dk * 32 + hi * 8);
      qf[dk].h[1] = *(const v8bf*)(qp + dk * 32 + 16 + hi * 8);
    }
  }

  const v8f vz = {0.f,0.f,0.f,0.f,0.f,0.f,0.f,0.f};
  v8f oacc[8];
#pragma unroll
  for (int i = 0; i < 8; i++) oacc[i] = vz;
  float mrow[8], lrow[8];
#pragma unroll
  for (int r = 0; r < 8; r++) { mrow[r] = -1e30f; lrow[r] = 0.0f; }

  auto stage = [&](int k0, int buf) {
#pragma unroll
    for (int i = 0; i < 4; i++) {            // K tile: 64 rows x 128 halfs
      int idx = tid + i * 256;
      int row = idx >> 4, seg = (idx & 15) * 8;
      async_copy_b128(lds_off(&lK[buf][row * KSTR + seg]),
                      &Kbase[(size_t)(k0 + row) * D_ + seg]);
    }
#pragma unroll
    for (int i = 0; i < 4; i++) {            // V^T tile: 128 rows x 64 halfs
      int idx = tid + i * 256;
      int row = idx >> 3, seg = (idx & 7) * 8;
      async_copy_b128(lds_off(&lV[buf][row * VSTR + seg]),
                      &Vbase[(size_t)row * S_ + k0 + seg]);
    }
  };

  const int nkt = (q0 + QT_) / KT_;           // causal bound, uniform per block
  stage(0, 0);
  for (int kt = 0; kt < nkt; kt++) {
    const int cur = kt & 1;
    const int k0 = kt * KT_;
    wait_async0();            // my DMAs for tile kt landed
    __syncthreads();          // tile kt visible to all; buffer !cur reusable

    if (kt + 1 < nkt) stage((kt + 1) * KT_, cur ^ 1);
    if (kt + 2 < nkt) {       // speculative L2 prefetch of tile kt+2
      const int k2 = (kt + 2) * KT_;
      if (tid < KT_) __builtin_prefetch(&Kbase[(size_t)(k2 + tid) * D_], 0, 0);
      if (tid < D_)  __builtin_prefetch(&Vbase[(size_t)tid * S_ + k2], 0, 0);
    }

    v8f sacc[4];
#pragma unroll
    for (int n = 0; n < 4; n++) sacc[n] = vz;
#pragma unroll
    for (int dk = 0; dk < 4; dk++)
#pragma unroll
      for (int n = 0; n < 4; n++) {
        V16 kf;
        const bf16* p = &lK[cur][(n * 16 + l16) * KSTR + dk * 32];
        kf.h[0] = *(const v8bf*)(p + hi * 8);
        kf.h[1] = *(const v8bf*)(p + 16 + hi * 8);
        sacc[n] = __builtin_amdgcn_wmma_f32_16x16x32_bf16(
            false, qf[dk].v, false, kf.v, (short)0, sacc[n], false, false);
      }

    const int rowg0 = q0 + wave * 16 + hi * 8;
#pragma unroll
    for (int n = 0; n < 4; n++) {
      const int colg = k0 + n * 16 + l16;
#pragma unroll
      for (int r = 0; r < 8; r++) {
        float sv = sacc[n][r] * SCALE_;
        sacc[n][r] = (colg <= rowg0 + r) ? sv : -1e30f;
      }
    }

    float alpha[8];
#pragma unroll
    for (int r = 0; r < 8; r++) {
      float mx = fmaxf(fmaxf(sacc[0][r], sacc[1][r]), fmaxf(sacc[2][r], sacc[3][r]));
#pragma unroll
      for (int m = 8; m >= 1; m >>= 1) mx = fmaxf(mx, __shfl_xor(mx, m, 32));
      const float mnew = fmaxf(mrow[r], mx);
      alpha[r] = __expf(mrow[r] - mnew);
      mrow[r] = mnew;
      float rs = 0.0f;
#pragma unroll
      for (int n = 0; n < 4; n++) {
        float p = __expf(sacc[n][r] - mnew);
        sacc[n][r] = p;
        rs += p;
      }
#pragma unroll
      for (int m = 8; m >= 1; m >>= 1) rs += __shfl_xor(rs, m, 32);
      lrow[r] = lrow[r] * alpha[r] + rs;
    }
#pragma unroll
    for (int i = 0; i < 8; i++)
#pragma unroll
      for (int r = 0; r < 8; r++) oacc[i][r] *= alpha[r];

    bf16* pw = lP[wave];      // per-wave buffer: intra-wave DS ordering only
#pragma unroll
    for (int n = 0; n < 4; n++)
#pragma unroll
      for (int r = 0; r < 8; r++)
        pw[(hi * 8 + r) * VSTR + n * 16 + l16] = f2bf(sacc[n][r]);

#pragma unroll
    for (int kk = 0; kk < 2; kk++) {
      V16 pf;
      const bf16* pp = pw + l16 * VSTR + kk * 32;
      pf.h[0] = *(const v8bf*)(pp + hi * 8);
      pf.h[1] = *(const v8bf*)(pp + 16 + hi * 8);
#pragma unroll
      for (int dn = 0; dn < 8; dn++) {
        V16 vf;
        const bf16* vp = &lV[cur][(dn * 16 + l16) * VSTR + kk * 32];
        vf.h[0] = *(const v8bf*)(vp + hi * 8);
        vf.h[1] = *(const v8bf*)(vp + 16 + hi * 8);
        oacc[dn] = __builtin_amdgcn_wmma_f32_16x16x32_bf16(
            false, pf.v, false, vf.v, (short)0, oacc[dn], false, false);
      }
    }
    // no trailing barrier: buffer `cur` is only rewritten after the next
    // iteration's barrier, which happens-after all reads above.
  }

#pragma unroll
  for (int dn = 0; dn < 8; dn++)
#pragma unroll
    for (int r = 0; r < 8; r++) {
      const int sq = q0 + wave * 16 + hi * 8 + r;
      const int col = h * D_ + dn * 16 + l16;
      Ab[((size_t)b * S_ + sq) * (size_t)(H_ * D_) + col] = f2bf(oacc[dn][r] / lrow[r]);
    }
}

// ---------------- host launcher ----------------
extern "C" void kernel_launch(void* const* d_in, const int* in_sizes, int n_in,
                              void* d_out, int out_size, void* d_ws, size_t ws_size,
                              hipStream_t stream) {
  (void)in_sizes; (void)n_in; (void)out_size; (void)ws_size;
  const int*   positions = (const int*)  d_in[0];
  const float* hidden    = (const float*)d_in[1];
  const float* Wq        = (const float*)d_in[2];
  const float* Wk        = (const float*)d_in[3];
  const float* Wv        = (const float*)d_in[4];
  const float* Wo        = (const float*)d_in[5];
  const float* qw        = (const float*)d_in[6];
  const float* kw        = (const float*)d_in[7];

  float* out_o = (float*)d_out;                        // [BS, HID]
  float* out_k = out_o + (size_t)BS_ * HID_;           // [BS, KVH*D]
  float* out_v = out_k + (size_t)BS_ * KVH_ * D_;      // [BS, KVH*D]

  char* w = (char*)d_ws; size_t off = 0;
  auto alloc = [&](size_t bytes) { void* p = w + off; off += (bytes + 255) & ~(size_t)255; return p; };
  bf16*  Xb  = (bf16*) alloc((size_t)BS_ * HID_ * 2);
  bf16*  WqT = (bf16*) alloc((size_t)HID_ * H_ * D_ * 2);
  bf16*  WkT = (bf16*) alloc((size_t)HID_ * KVH_ * D_ * 2);
  bf16*  WvT = (bf16*) alloc((size_t)HID_ * KVH_ * D_ * 2);
  bf16*  WoT = (bf16*) alloc((size_t)H_ * D_ * HID_ * 2);
  float* Qf  = (float*)alloc((size_t)BS_ * H_ * D_ * 4);
  float* Kf  = (float*)alloc((size_t)BS_ * KVH_ * D_ * 4);
  float* Vf  = (float*)alloc((size_t)BS_ * KVH_ * D_ * 4);
  // buffer reuse (stream-ordered, producers of the old contents are done):
  bf16* Qb = Xb;    // written after last GEMM read of Xb
  bf16* Kb = WkT;   // written after K-projection GEMM
  bf16* Vt = WvT;   // written after V-projection GEMM
  bf16* Ab = WqT;   // written after Q-projection GEMM

  // 0) precision conversion + LDS-tiled weight transposes (NT GEMM layout)
  { int n = BS_ * HID_;
    f32_to_bf16_kernel<<<dim3((n / 4 + 255) / 256), dim3(256), 0, stream>>>(hidden, Xb, n); }
  transpose_to_bf16_kernel<<<dim3((HID_ / 32) * ((H_ * D_) / 32)), dim3(256), 0, stream>>>(Wq, WqT, HID_, H_ * D_);
  transpose_to_bf16_kernel<<<dim3((HID_ / 32) * ((KVH_ * D_) / 32)), dim3(256), 0, stream>>>(Wk, WkT, HID_, KVH_ * D_);
  transpose_to_bf16_kernel<<<dim3((HID_ / 32) * ((KVH_ * D_) / 32)), dim3(256), 0, stream>>>(Wv, WvT, HID_, KVH_ * D_);
  transpose_to_bf16_kernel<<<dim3(((H_ * D_) / 32) * (HID_ / 32)), dim3(256), 0, stream>>>(Wo, WoT, H_ * D_, HID_);

  // 1) QKV projections (bf16 WMMA, f32 accumulate, async-LDS staged)
  gemm_nt_bf16<<<dim3((H_ * D_) / TN_, BS_ / TM_), dim3(256), 0, stream>>>(Xb, WqT, Qf, BS_, H_ * D_, HID_);
  gemm_nt_bf16<<<dim3((KVH_ * D_) / TN_, BS_ / TM_), dim3(256), 0, stream>>>(Xb, WkT, Kf, BS_, KVH_ * D_, HID_);
  gemm_nt_bf16<<<dim3((KVH_ * D_) / TN_, BS_ / TM_), dim3(256), 0, stream>>>(Xb, WvT, Vf, BS_, KVH_ * D_, HID_);

  // 2a) RMSNorm + RoPE for q/k heads; emits k output + bf16 Q/K
  rmsnorm_rope_pack<<<dim3(BS_), dim3(256), 0, stream>>>(
      positions, Qf, Kf, qw, kw, Qb, Kb, out_k);
  // 2b) V: coalesced f32 copy-out + LDS-tiled transpose to bf16 V^T
  v_pack_transpose_kernel<<<dim3((S_ / 32) * (D_ / 32), B_ * KVH_), dim3(256), 0, stream>>>(
      Vf, out_v, Vt);

  // 3) causal flash attention (WMMA for QK^T and PV, async-LDS staged K/V)
  flash_attn_kernel<<<dim3(S_ / QT_, B_ * H_), dim3(256), 0, stream>>>(Qb, Kb, Vt, Ab);

  // 4) output projection
  gemm_nt_bf16<<<dim3(HID_ / TN_, BS_ / TM_), dim3(256), 0, stream>>>(Ab, WoT, out_o, BS_, HID_, H_ * D_);
}
